// HTAttention_39986145526411
// MI455X (gfx1250) — matrix operand order repaired and checked by reading, not compiled, not run on hardware
//
#include <hip/hip_runtime.h>
#include <hip/hip_bf16.h>

// ---------------------------------------------------------------------------
// HTAttention (linear attention) for MI455X / gfx1250, wave32 + bf16 WMMA.
//
//   Q = x wq^T ; K = x wk^T ; V = x wv^T        (bf16 WMMA, f32 accum)
//   energy[b] = K^T V                           (bf16 WMMA, f32 accum)
//   M[b] = scale * energy[b] @ wo^T             (tiny, scalar f32)
//   out  = Q @ M                                (bf16 WMMA, f32 out)
//
// Memory-bound: ~280 MB HBM traffic (x read + out write) -> ~12 us floor at
// 23.3 TB/s; 17.7 GFLOP of WMMA is negligible next to that, so the design
// reads x exactly once and keeps Q/KT/VT/MT as bf16 in L2 (~13 MB workspace).
//
// v2: projection loop software-pipelines the x stream (double-buffered fp32
// chunk + global_prefetch two chunks past the load) and preloads all 12 B-frags into
// registers before the 12 WMMAs so loads and WMMAs overlap instead of the
// per-WMMA `s_wait_loadcnt 0` serialization seen in v1 codegen.
// ---------------------------------------------------------------------------

typedef __attribute__((ext_vector_type(16))) __bf16 v16bf;
typedef __attribute__((ext_vector_type(8)))  __bf16 v8bf;
typedef __attribute__((ext_vector_type(8)))  float  v8f;
typedef __attribute__((ext_vector_type(4)))  float  v4f;

#define B_    4
#define N_    8192
#define E_    1024
#define D_    64
#define ROWS_ (B_ * N_)          // 32768
#define SCALE_ 0.21650635094610965f   // sqrt(3/64)

// Workspace layout (bytes)
#define WS_Q      0                            // 32768*64 bf16   = 4 MiB
#define WS_KT     (4194304)                    // [b][d][n] bf16  = 4 MiB
#define WS_VT     (8388608)                    // [b][d][n] bf16  = 4 MiB
#define WS_ENERGY (12582912)                   // [b][64][64] f32 = 64 KiB
#define WS_MT     (12648448)                   // [b][e][d] bf16  = 512 KiB
#define WS_WBF    (13172736)                   // [3][64][1024] bf16 = 384 KiB
#define WS_PART   (13565952)                   // 256 wave-partials [64x64] f32 = 4 MiB

static __device__ __forceinline__ v16bf cat16(v8bf lo, v8bf hi) {
  v16bf r;
#pragma unroll
  for (int i = 0; i < 8; ++i) { r[i] = lo[i]; r[i + 8] = hi[i]; }
  return r;
}

static __device__ __forceinline__ v8bf cvt8(v4f a, v4f b) {
  v8bf r;
#pragma unroll
  for (int i = 0; i < 4; ++i) { r[i] = (__bf16)a[i]; r[i + 4] = (__bf16)b[i]; }
  return r;
}

static __device__ __forceinline__ v16bf load_a16(const __bf16* p) {
  // A-frag lane layout: K chunks [8h..8h+7] and [16+8h..16+8h+7] (caller adds 8h)
  return cat16(*(const v8bf*)p, *(const v8bf*)(p + 16));
}
static __device__ __forceinline__ v16bf load_b16(const __bf16* p) {
  // B-frag lane layout: K chunks [8h..8h+7] and [16+8h..16+8h+7] (caller adds 8h)
  return cat16(*(const v8bf*)p, *(const v8bf*)(p + 16));
}

static __device__ __forceinline__ v8f wmma_bf16(v16bf a, v16bf b, v8f c) {
  return __builtin_amdgcn_wmma_f32_16x16x32_bf16(false, a, false, b,
                                                 (short)0, c, false, false);
}

// --------------------------------------------------------------------------
// Kernel 1: convert wq/wk/wv (fp32, [64][1024] each) to bf16 [3][64][1024]
// --------------------------------------------------------------------------
__global__ __launch_bounds__(256) void convw_kernel(
    const float* __restrict__ wq, const float* __restrict__ wk,
    const float* __restrict__ wv, __bf16* __restrict__ wbf) {
  int i = blockIdx.x * 256 + threadIdx.x;               // < 3*65536
  const float* src = (i < 65536) ? wq : ((i < 131072) ? wk : wv);
  wbf[i] = (__bf16)src[i & 65535];
}

// --------------------------------------------------------------------------
// Kernel 2: fused projections.  Block = 256 thr (8 waves), 128 rows/block.
// Each wave owns 16 rows; the x A-frag is converted to bf16 in-register and
// reused for all 3 projections x 4 n-tiles = 12 WMMAs per 32-wide k-step.
// x chunk for step kb+32 is fetched while step kb's WMMAs run; all 12 B-frags
// are resident in registers before the WMMA burst.
// Writes Q row-major bf16, and K,V *transposed* ([b][d][n]) for the energy GEMM.
// --------------------------------------------------------------------------
__global__ __launch_bounds__(256) void proj_kernel(
    const float* __restrict__ x, const __bf16* __restrict__ wbf,
    __bf16* __restrict__ Q, __bf16* __restrict__ KT, __bf16* __restrict__ VT) {
  const int wave = threadIdx.x >> 5, lane = threadIdx.x & 31;
  const int m = lane & 15, h = lane >> 4;
  const int row0 = blockIdx.x * 128 + wave * 16;

  v8f accQ[4], accK[4], accV[4];
#pragma unroll
  for (int t = 0; t < 4; ++t) { accQ[t] = v8f{}; accK[t] = v8f{}; accV[t] = v8f{}; }

  const float* xr = x + (size_t)(row0 + m) * E_ + 8 * h;

  // Prologue: stage first x chunk (16 floats: K = [8h..8h+7], [16+8h..+7])
  v4f x0 = *(const v4f*)(xr + 0);
  v4f x1 = *(const v4f*)(xr + 4);
  v4f x2 = *(const v4f*)(xr + 16);
  v4f x3 = *(const v4f*)(xr + 20);

  for (int kb = 0; kb < E_; kb += 32) {
    v16bf a = cat16(cvt8(x0, x1), cvt8(x2, x3));

    if (kb + 32 < E_) {                       // uniform branch (SCC)
      const float* pn = xr + kb + 32;
      __builtin_prefetch(pn + 64, 0, 1);      // global_prefetch two chunks past this load
      x0 = *(const v4f*)(pn + 0);
      x1 = *(const v4f*)(pn + 4);
      x2 = *(const v4f*)(pn + 16);
      x3 = *(const v4f*)(pn + 20);
    }

    // Preload all 12 B-frags (wq/wk/wv rows, L2-hot), then burst 12 WMMAs.
    const __bf16* wb = wbf + (size_t)m * E_ + kb + 8 * h;
    v16bf bfr[12];
#pragma unroll
    for (int t = 0; t < 4; ++t) {
      const __bf16* w0 = wb + (size_t)(t * 16) * E_;
      bfr[t]     = load_b16(w0);                        // wq
      bfr[4 + t] = load_b16(w0 + (size_t)64 * E_);      // wk
      bfr[8 + t] = load_b16(w0 + (size_t)128 * E_);     // wv
    }
#pragma unroll
    for (int t = 0; t < 4; ++t) {
      accQ[t] = wmma_bf16(a, bfr[t],     accQ[t]);
      accK[t] = wmma_bf16(a, bfr[4 + t], accK[t]);
      accV[t] = wmma_bf16(a, bfr[8 + t], accV[t]);
    }
  }

  // C layout: VGPR j, lane(h,n') -> row = j + 8h, col = n' = lane&15
#pragma unroll
  for (int t = 0; t < 4; ++t) {
#pragma unroll
    for (int j = 0; j < 8; ++j) {
      const int r = row0 + j + 8 * h;
      const int c = t * 16 + m;
      Q[(size_t)r * D_ + c] = (__bf16)accQ[t][j];
      const int b = r >> 13, n = r & (N_ - 1);
      const size_t ti = ((size_t)(b * D_ + c) << 13) + n;   // [b][d][n]
      KT[ti] = (__bf16)accK[t][j];
      VT[ti] = (__bf16)accV[t][j];
    }
  }
}

// --------------------------------------------------------------------------
// Kernel 3: energy partials.  energy[b][d][e] = sum_n K[n,d] V[n,e].
// A = KT rows (contiguous in n), B = VT rows (contiguous in n).
// Grid = (8 chunks, B); each of the 8 waves reduces its own 128-wide n slice
// into a private [64x64] accumulator, stored as a wave-partial (deterministic).
// --------------------------------------------------------------------------
__global__ __launch_bounds__(256) void energy_kernel(
    const __bf16* __restrict__ KT, const __bf16* __restrict__ VT,
    float* __restrict__ part) {
  const int b = blockIdx.y;
  const int wave = threadIdx.x >> 5, lane = threadIdx.x & 31;
  const int m = lane & 15, h = lane >> 4;
  const int nbase = blockIdx.x * 1024 + wave * 128;
  const __bf16* KTb = KT + ((size_t)b * D_ << 13);
  const __bf16* VTb = VT + ((size_t)b * D_ << 13);

  v8f acc[4][4];
#pragma unroll
  for (int i = 0; i < 4; ++i)
#pragma unroll
    for (int j = 0; j < 4; ++j) acc[i][j] = v8f{};

#pragma unroll
  for (int ks = 0; ks < 4; ++ks) {
    const int n0 = nbase + ks * 32;
    v16bf af[4], bfr[4];
#pragma unroll
    for (int dt = 0; dt < 4; ++dt)
      af[dt] = load_a16(KTb + ((size_t)(dt * 16 + m) << 13) + n0 + 8 * h);
#pragma unroll
    for (int et = 0; et < 4; ++et)
      bfr[et] = load_b16(VTb + ((size_t)(et * 16 + m) << 13) + n0 + 8 * h);
#pragma unroll
    for (int dt = 0; dt < 4; ++dt)
#pragma unroll
      for (int et = 0; et < 4; ++et)
        acc[dt][et] = wmma_bf16(af[dt], bfr[et], acc[dt][et]);
  }

  float* wp = part + (size_t)((b * 8 + blockIdx.x) * 8 + wave) * 4096;
#pragma unroll
  for (int dt = 0; dt < 4; ++dt)
#pragma unroll
    for (int et = 0; et < 4; ++et)
#pragma unroll
      for (int j = 0; j < 8; ++j)
        wp[(dt * 16 + j + 8 * h) * 64 + et * 16 + m] = acc[dt][et][j];
}

// Kernel 4: reduce 64 wave-partials per batch in fixed order (deterministic).
__global__ __launch_bounds__(256) void reduce_kernel(
    const float* __restrict__ part, float* __restrict__ energy) {
  const int idx = blockIdx.x * 256 + threadIdx.x;       // < B*4096
  const int b = idx >> 12, i = idx & 4095;
  const float* p = part + (size_t)b * 64 * 4096 + i;
  float s = 0.f;
#pragma unroll
  for (int w = 0; w < 64; ++w) s += p[(size_t)w * 4096];
  energy[idx] = s;
}

// --------------------------------------------------------------------------
// Kernel 5: MT[b][e][d1] = scale * sum_d2 energy[b][d1][d2] * wo[e][d2]
// (stored transposed so the final GEMM's B-frag reads are contiguous)
// --------------------------------------------------------------------------
__global__ __launch_bounds__(256) void mt_kernel(
    const float* __restrict__ energy, const float* __restrict__ wo,
    __bf16* __restrict__ MT) {
  const int idx = blockIdx.x * 256 + threadIdx.x;       // < B*E*D = 262144
  const int b = idx >> 16, rem = idx & 65535, e = rem >> 6, d1 = rem & 63;
  const float* er = energy + b * 4096 + d1 * 64;
  const float* wr = wo + e * 64;
  float s = 0.f;
#pragma unroll
  for (int d2 = 0; d2 < 64; ++d2) s += er[d2] * wr[d2];
  MT[((size_t)b << 16) + e * 64 + d1] = (__bf16)(s * SCALE_);
}

// --------------------------------------------------------------------------
// Kernel 6: out[b][n][e] = sum_d Q[b][n][d] * MT[b][e][d]   (f32 out)
// Block tile 64 rows x 256 cols; wave grid 4x2 (16 rows x 128 cols / wave).
// --------------------------------------------------------------------------
__global__ __launch_bounds__(256) void out_kernel(
    const __bf16* __restrict__ Q, const __bf16* __restrict__ MT,
    float* __restrict__ out) {
  const int wave = threadIdx.x >> 5, lane = threadIdx.x & 31;
  const int m = lane & 15, h = lane >> 4;
  const int r0 = blockIdx.x * 64 + (wave & 3) * 16;
  const int c0 = blockIdx.y * 256 + (wave >> 2) * 128;
  const int b = (blockIdx.x * 64) >> 13;                 // 8192 % 64 == 0
  const __bf16* MTb = MT + ((size_t)b << 16);

  v8f acc[8];
#pragma unroll
  for (int t = 0; t < 8; ++t) acc[t] = v8f{};

#pragma unroll
  for (int ks = 0; ks < 2; ++ks) {
    const int k0 = ks * 32;
    v16bf a = load_a16(Q + (size_t)(r0 + m) * D_ + k0 + 8 * h);
    v16bf bfr[8];
#pragma unroll
    for (int t = 0; t < 8; ++t)
      bfr[t] = load_b16(MTb + (size_t)(c0 + t * 16 + m) * D_ + k0 + 8 * h);
#pragma unroll
    for (int t = 0; t < 8; ++t)
      acc[t] = wmma_bf16(a, bfr[t], acc[t]);
  }

#pragma unroll
  for (int t = 0; t < 8; ++t)
#pragma unroll
    for (int j = 0; j < 8; ++j)
      out[(size_t)(r0 + j + 8 * h) * E_ + c0 + t * 16 + m] = acc[t][j];
}

extern "C" void kernel_launch(void* const* d_in, const int* in_sizes, int n_in,
                              void* d_out, int out_size, void* d_ws, size_t ws_size,
                              hipStream_t stream) {
  (void)in_sizes; (void)n_in; (void)out_size; (void)ws_size;
  const float* x  = (const float*)d_in[0];
  const float* wq = (const float*)d_in[1];
  const float* wk = (const float*)d_in[2];
  const float* wv = (const float*)d_in[3];
  const float* wo = (const float*)d_in[4];
  float* out = (float*)d_out;

  char* ws = (char*)d_ws;
  __bf16* Q      = (__bf16*)(ws + WS_Q);
  __bf16* KT     = (__bf16*)(ws + WS_KT);
  __bf16* VT     = (__bf16*)(ws + WS_VT);
  float*  energy = (float*)(ws + WS_ENERGY);
  __bf16* MT     = (__bf16*)(ws + WS_MT);
  __bf16* wbf    = (__bf16*)(ws + WS_WBF);
  float*  part   = (float*)(ws + WS_PART);

  convw_kernel<<<768, 256, 0, stream>>>(wq, wk, wv, wbf);
  proj_kernel<<<ROWS_ / 128, 256, 0, stream>>>(x, wbf, Q, KT, VT);
  energy_kernel<<<dim3(8, B_), 256, 0, stream>>>(KT, VT, part);
  reduce_kernel<<<(B_ * 4096) / 256, 256, 0, stream>>>(part, energy);
  mt_kernel<<<(B_ * E_ * D_) / 256, 256, 0, stream>>>(energy, wo, MT);
  out_kernel<<<dim3(ROWS_ / 64, E_ / 256), 256, 0, stream>>>(Q, MT, out);
}
